// MultiHeadAttention_8959301779696
// MI455X (gfx1250) — matrix operand order repaired and checked by reading, not compiled, not run on hardware
//
#include <hip/hip_runtime.h>
#include <hip/hip_bf16.h>
#include <stdint.h>

#define NUM_HEAD 16
#define EMBED    1024
#define HEAD_DIM 64
#define BATCH    8
#define SEQ      1024

typedef __attribute__((ext_vector_type(16))) __bf16 v16bf;
typedef __attribute__((ext_vector_type(8)))  float  v8f;

union FragBF { v16bf v; uint4 q[2]; };
union AccF   { v8f   v; float f[8]; };

__device__ __forceinline__ unsigned short f2bf(float x) {
    uint32_t u = __float_as_uint(x);
    u += 0x7FFFu + ((u >> 16) & 1u);           // round-to-nearest-even
    return (unsigned short)(u >> 16);
}
__device__ __forceinline__ uint32_t pack2bf(float lo, float hi) {
    return (uint32_t)f2bf(lo) | ((uint32_t)f2bf(hi) << 16);
}
__device__ __forceinline__ v16bf ld_frag(const unsigned short* p0,
                                         const unsigned short* p1) {
    FragBF f;
    f.q[0] = *(const uint4*)p0;
    f.q[1] = *(const uint4*)p1;
    return f.v;
}
__device__ __forceinline__ v8f wmma_bf16(v16bf a, v16bf b, v8f c) {
    return __builtin_amdgcn_wmma_f32_16x16x32_bf16(false, a, false, b,
                                                   (short)0, c, false, false);
}

// ---- gfx1250 async global->LDS copy (ASYNCcnt) with safe fallback ----------
#if defined(__AMDGCN__) && defined(__gfx1250__) && \
    __has_builtin(__builtin_amdgcn_global_load_async_to_lds_b128) && \
    __has_builtin(__builtin_amdgcn_s_wait_asynccnt)
#define HAVE_ASYNC 1
#else
#define HAVE_ASYNC 0
#endif

#if HAVE_ASYNC
// Parameter type per hipcc diagnostic: AS(1) pointer to int vector_size(16).
typedef int v4i_vs __attribute__((vector_size(16)));
typedef __attribute__((address_space(1))) v4i_vs* gv4_ptr;
typedef __attribute__((address_space(3))) v4i_vs* lv4_ptr;
__device__ __forceinline__ void async_cp16(const unsigned short* g, unsigned short* l) {
    __builtin_amdgcn_global_load_async_to_lds_b128((gv4_ptr)(g), (lv4_ptr)(l), 0, 0);
}
#endif

// Prefetch 32 bf16 elements (64B): async issues the DMA, fallback stages regs.
__device__ __forceinline__ void stage_prefetch(const unsigned short* g,
                                               unsigned short* l, uint4* regs) {
#if HAVE_ASYNC
    (void)regs;
    #pragma unroll
    for (int i = 0; i < 4; i++) async_cp16(g + i * 8, l + i * 8);
#else
    (void)l;
    #pragma unroll
    for (int i = 0; i < 4; i++) regs[i] = ((const uint4*)g)[i];
#endif
}
__device__ __forceinline__ void stage_commit(unsigned short* l, const uint4* regs) {
#if !HAVE_ASYNC
    #pragma unroll
    for (int i = 0; i < 4; i++) ((uint4*)l)[i] = regs[i];
#else
    (void)l; (void)regs;
#endif
}
__device__ __forceinline__ void stage_wait() {
#if HAVE_ASYNC
    __builtin_amdgcn_s_wait_asynccnt(0);
#endif
}

// ---------------------------------------------------------------------------
// Kernel 0: fp32 -> bf16 bulk convert (8 elements / thread)
// ---------------------------------------------------------------------------
__global__ __launch_bounds__(256) void cvt_bf16_kernel(const float* __restrict__ src,
                                                       unsigned short* __restrict__ dst) {
    size_t i = ((size_t)blockIdx.x * 256 + threadIdx.x) * 8;
    const float4* s = (const float4*)(src + i);
    float4 f0 = s[0], f1 = s[1];
    uint4 o;
    o.x = pack2bf(f0.x, f0.y); o.y = pack2bf(f0.z, f0.w);
    o.z = pack2bf(f1.x, f1.y); o.w = pack2bf(f1.z, f1.w);
    *(uint4*)(dst + i) = o;
}

// ---------------------------------------------------------------------------
// GEMM geometry: 128x128 tile/block, 8 waves (4x2), 32x64 per wave, K-tile 64.
// ---------------------------------------------------------------------------
#define MT  128
#define NT  128
#define KT2 64
#define LDK 72    // padded LDS stride (elements), keeps 16B alignment
#define LV  136   // padded 128-wide stride

#define QKV_ELEMS ((size_t)BATCH * NUM_HEAD * SEQ * HEAD_DIM)   // 8Mi

// ---------------------------------------------------------------------------
// Kernel 1: QKV projection (bf16 in, bf16 [3][B,H,S,D] out, branch-free scatter)
// ---------------------------------------------------------------------------
__global__ __launch_bounds__(256) void qkv_gemm_kernel(
    const unsigned short* __restrict__ xb, const unsigned short* __restrict__ wb,
    const float* __restrict__ bias, unsigned short* __restrict__ qkv)
{
    __shared__ unsigned short As[2][MT * LDK];
    __shared__ unsigned short Bs[2][NT * LDK];

    const int tid  = threadIdx.x;
    const int lane = tid & 31, wid = tid >> 5;
    const int wm = wid >> 1, wn = wid & 1;
    const int m0 = blockIdx.x * MT;
    const int n0 = blockIdx.y * NT;

    const int lrow = tid >> 1;
    const int lseg = (tid & 1) * 32;
    const unsigned short* ga = xb + (size_t)(m0 + lrow) * EMBED + lseg;
    const unsigned short* gb = wb + (size_t)(n0 + lrow) * EMBED + lseg;
    const int lofs = lrow * LDK + lseg;

    AccF acc[2][4];
    #pragma unroll
    for (int i = 0; i < 2; i++)
        #pragma unroll
        for (int j = 0; j < 4; j++)
            #pragma unroll
            for (int r = 0; r < 8; r++) acc[i][j].f[r] = 0.0f;

    const int arow = lane & 15;
    const int kb8  = (lane >> 4) * 8;
    const int kb16 = (lane >> 4) * 16;

    uint4 sa[4], sb[4];
    stage_prefetch(ga, &As[0][lofs], sa);
    stage_prefetch(gb, &Bs[0][lofs], sb);

    #pragma unroll 2
    for (int kt = 0; kt < EMBED / KT2; ++kt) {
        const int buf = kt & 1;
        stage_wait();
        stage_commit(&As[buf][lofs], sa);
        stage_commit(&Bs[buf][lofs], sb);
        __syncthreads();
        if (kt + 1 < EMBED / KT2) {
            stage_prefetch(ga + (kt + 1) * KT2, &As[buf ^ 1][lofs], sa);
            stage_prefetch(gb + (kt + 1) * KT2, &Bs[buf ^ 1][lofs], sb);
        }
        #pragma unroll
        for (int ks = 0; ks < 2; ++ks) {
            const unsigned short* a0p = &As[buf][(wm * 32 + arow) * LDK + ks * 32 + kb8];
            const unsigned short* a1p = a0p + 16 * LDK;
            v16bf a0 = ld_frag(a0p, a0p + 16);
            v16bf a1 = ld_frag(a1p, a1p + 16);
            #pragma unroll
            for (int j = 0; j < 4; j++) {
                const unsigned short* bp =
                    &Bs[buf][(wn * 64 + j * 16 + arow) * LDK + ks * 32 + kb8];
                v16bf b = ld_frag(bp, bp + 16);
                acc[0][j].v = wmma_bf16(a0, b, acc[0][j].v);
                acc[1][j].v = wmma_bf16(a1, b, acc[1][j].v);
            }
        }
    }

    // branch-free epilogue: bias, fold 1/sqrt(D) into Q, scatter [which][B,H,S,D]
    const int laneHi = lane >> 4;
    const int lanN   = lane & 15;
    #pragma unroll
    for (int i = 0; i < 2; i++) {
        #pragma unroll
        for (int j = 0; j < 4; j++) {
            #pragma unroll
            for (int r = 0; r < 8; r++) {
                int m = m0 + wm * 32 + i * 16 + r + laneHi * 8;
                int n = n0 + wn * 64 + j * 16 + lanN;
                float v = acc[i][j].f[r] + bias[n];
                int which = n >> 10;           // 0=Q 1=K 2=V
                int hc = n & 1023;
                int h = hc >> 6, d = hc & 63;
                int b = m >> 10, s = m & 1023;
                size_t off = (size_t)which * QKV_ELEMS +
                             (((size_t)(b * NUM_HEAD + h) * SEQ) + s) * HEAD_DIM + d;
                float scale = (which == 0) ? 0.125f : 1.0f;
                qkv[off] = f2bf(v * scale);
            }
        }
    }
}

// ---------------------------------------------------------------------------
// Kernel 2: flash attention, double-buffered K/V tiles.
// ---------------------------------------------------------------------------
#define QT  128
#define KVT 128

__global__ __launch_bounds__(256) void attn_kernel(
    const unsigned short* __restrict__ Q, const unsigned short* __restrict__ K,
    const unsigned short* __restrict__ V, unsigned short* __restrict__ ctx)
{
    __shared__ unsigned short Qs[QT * LDK];
    __shared__ unsigned short Ks[2][KVT * LDK];
    __shared__ unsigned short Vt[2][HEAD_DIM * LV];   // V transposed [d][key]
    __shared__ unsigned short Ps[QT * LV];            // P tile (wave-local rows)

    const int tid  = threadIdx.x;
    const int lane = tid & 31, wid = tid >> 5;
    const int bh = blockIdx.y;
    const int q0 = blockIdx.x * QT;
    const size_t base = (size_t)bh * SEQ * HEAD_DIM;

    const int lrow = tid >> 1;
    const int lseg = (tid & 1) * 32;
    const int lofs = lrow * LDK + lseg;

    // prefetch KV tile 0 while staging Q
    uint4 kregs[4], vregs[4];
    stage_prefetch(K + base + (size_t)lrow * HEAD_DIM + lseg, &Ks[0][lofs], kregs);
    #pragma unroll
    for (int i = 0; i < 4; i++)
        vregs[i] = ((const uint4*)(V + base + (size_t)lrow * HEAD_DIM + lseg))[i];
    {
        const uint4* src = (const uint4*)(Q + base + (size_t)(q0 + lrow) * HEAD_DIM + lseg);
        uint4* dst = (uint4*)(Qs + lofs);
        dst[0] = src[0]; dst[1] = src[1]; dst[2] = src[2]; dst[3] = src[3];
    }
    __syncthreads();

    const int arow  = lane & 15;
    const int kb8   = (lane >> 4) * 8;
    const int kb16  = (lane >> 4) * 16;
    const int myrow = wid * 16 + arow;

    v16bf qa[2];
    #pragma unroll
    for (int s = 0; s < 2; s++) {
        const unsigned short* p = Qs + myrow * LDK + s * 32 + kb8;
        qa[s] = ld_frag(p, p + 16);
    }

    float m_run[8], l_run[8];
    AccF o[4];
    #pragma unroll
    for (int r = 0; r < 8; r++) { m_run[r] = -INFINITY; l_run[r] = 0.0f; }
    #pragma unroll
    for (int n = 0; n < 4; n++)
        #pragma unroll
        for (int r = 0; r < 8; r++) o[n].f[r] = 0.0f;

    #pragma unroll 2
    for (int t = 0; t < SEQ / KVT; ++t) {
        const int buf = t & 1;
        stage_wait();
        stage_commit(&Ks[buf][lofs], kregs);
        {   // commit V transposed: Vt[d][key]
            const uint32_t* uu = (const uint32_t*)vregs;
            #pragma unroll
            for (int e = 0; e < 16; e++) {
                int d = lseg + e * 2;
                uint32_t u = uu[e];
                Vt[buf][d * LV + lrow]       = (unsigned short)(u & 0xFFFFu);
                Vt[buf][(d + 1) * LV + lrow] = (unsigned short)(u >> 16);
            }
        }
        __syncthreads();
        if (t + 1 < SEQ / KVT) {
            const size_t nb = base + (size_t)((t + 1) * KVT + lrow) * HEAD_DIM + lseg;
            stage_prefetch(K + nb, &Ks[buf ^ 1][lofs], kregs);
            #pragma unroll
            for (int i = 0; i < 4; i++) vregs[i] = ((const uint4*)(V + nb))[i];
        }

        // S = Q K^T  (16 rows x 128 keys per wave)
        AccF s[8];
        #pragma unroll
        for (int j = 0; j < 8; j++) {
            #pragma unroll
            for (int r = 0; r < 8; r++) s[j].f[r] = 0.0f;
            #pragma unroll
            for (int ks = 0; ks < 2; ks++) {
                const unsigned short* bp =
                    &Ks[buf][(j * 16 + arow) * LDK + ks * 32 + kb8];
                v16bf kb = ld_frag(bp, bp + 16);
                s[j].v = wmma_bf16(qa[ks], kb, s[j].v);
            }
        }

        // online softmax (row = r + 8*(lane>=16); reduce over 16-lane halves)
        #pragma unroll
        for (int r = 0; r < 8; r++) {
            float mcur = s[0].f[r];
            #pragma unroll
            for (int j = 1; j < 8; j++) mcur = fmaxf(mcur, s[j].f[r]);
            #pragma unroll
            for (int msk = 1; msk < 16; msk <<= 1)
                mcur = fmaxf(mcur, __shfl_xor(mcur, msk, 32));
            float mnew = fmaxf(m_run[r], mcur);
            float corr = __expf(m_run[r] - mnew);
            m_run[r] = mnew;
            float psum = 0.0f;
            #pragma unroll
            for (int j = 0; j < 8; j++) {
                float p = __expf(s[j].f[r] - mnew);
                s[j].f[r] = p;
                psum += p;
            }
            #pragma unroll
            for (int msk = 1; msk < 16; msk <<= 1)
                psum += __shfl_xor(psum, msk, 32);
            l_run[r] = l_run[r] * corr + psum;
            #pragma unroll
            for (int n = 0; n < 4; n++) o[n].f[r] *= corr;
        }

        // spill P (C-layout) to LDS bf16; wave-local rows, DS in-order
        #pragma unroll
        for (int j = 0; j < 8; j++)
            #pragma unroll
            for (int r = 0; r < 8; r++)
                Ps[(wid * 16 + r + (lane >> 4) * 8) * LV + j * 16 + (lane & 15)]
                    = f2bf(s[j].f[r]);

        // O += P @ V
        #pragma unroll
        for (int ks = 0; ks < 4; ks++) {
            const unsigned short* pp = Ps + myrow * LV + ks * 32 + kb8;
            v16bf pa = ld_frag(pp, pp + 16);
            #pragma unroll
            for (int n = 0; n < 4; n++) {
                const unsigned short* vpp =
                    &Vt[buf][(n * 16 + arow) * LV + ks * 32 + kb8];
                v16bf vb = ld_frag(vpp, vpp + 16);
                o[n].v = wmma_bf16(pa, vb, o[n].v);
            }
        }
    }

    // epilogue: ctx[b, s, h*64 + d] bf16 ([B*S, E] layout for proj GEMM)
    const int h = bh & (NUM_HEAD - 1);
    const int b = bh >> 4;
    #pragma unroll
    for (int n = 0; n < 4; n++) {
        #pragma unroll
        for (int r = 0; r < 8; r++) {
            int srow = q0 + wid * 16 + r + (lane >> 4) * 8;
            int d = n * 16 + (lane & 15);
            float v = o[n].f[r] / l_run[r];
            ctx[((size_t)(b * SEQ + srow)) * EMBED + h * HEAD_DIM + d] = f2bf(v);
        }
    }
}

// ---------------------------------------------------------------------------
// Kernel 3: output projection (bf16 in, fp32 out + bias)
// ---------------------------------------------------------------------------
__global__ __launch_bounds__(256) void proj_gemm_kernel(
    const unsigned short* __restrict__ ctx, const unsigned short* __restrict__ wb,
    const float* __restrict__ bias, float* __restrict__ out)
{
    __shared__ unsigned short As[2][MT * LDK];
    __shared__ unsigned short Bs[2][NT * LDK];

    const int tid  = threadIdx.x;
    const int lane = tid & 31, wid = tid >> 5;
    const int wm = wid >> 1, wn = wid & 1;
    const int m0 = blockIdx.x * MT;
    const int n0 = blockIdx.y * NT;

    const int lrow = tid >> 1;
    const int lseg = (tid & 1) * 32;
    const unsigned short* ga = ctx + (size_t)(m0 + lrow) * EMBED + lseg;
    const unsigned short* gb = wb + (size_t)(n0 + lrow) * EMBED + lseg;
    const int lofs = lrow * LDK + lseg;

    AccF acc[2][4];
    #pragma unroll
    for (int i = 0; i < 2; i++)
        #pragma unroll
        for (int j = 0; j < 4; j++)
            #pragma unroll
            for (int r = 0; r < 8; r++) acc[i][j].f[r] = 0.0f;

    const int arow = lane & 15;
    const int kb8  = (lane >> 4) * 8;
    const int kb16 = (lane >> 4) * 16;

    uint4 sa[4], sb[4];
    stage_prefetch(ga, &As[0][lofs], sa);
    stage_prefetch(gb, &Bs[0][lofs], sb);

    #pragma unroll 2
    for (int kt = 0; kt < EMBED / KT2; ++kt) {
        const int buf = kt & 1;
        stage_wait();
        stage_commit(&As[buf][lofs], sa);
        stage_commit(&Bs[buf][lofs], sb);
        __syncthreads();
        if (kt + 1 < EMBED / KT2) {
            stage_prefetch(ga + (kt + 1) * KT2, &As[buf ^ 1][lofs], sa);
            stage_prefetch(gb + (kt + 1) * KT2, &Bs[buf ^ 1][lofs], sb);
        }
        #pragma unroll
        for (int ks = 0; ks < 2; ++ks) {
            const unsigned short* a0p = &As[buf][(wm * 32 + arow) * LDK + ks * 32 + kb8];
            const unsigned short* a1p = a0p + 16 * LDK;
            v16bf a0 = ld_frag(a0p, a0p + 16);
            v16bf a1 = ld_frag(a1p, a1p + 16);
            #pragma unroll
            for (int j = 0; j < 4; j++) {
                const unsigned short* bp =
                    &Bs[buf][(wn * 64 + j * 16 + arow) * LDK + ks * 32 + kb8];
                v16bf b = ld_frag(bp, bp + 16);
                acc[0][j].v = wmma_bf16(a0, b, acc[0][j].v);
                acc[1][j].v = wmma_bf16(a1, b, acc[1][j].v);
            }
        }
    }

    const int laneHi = lane >> 4;
    const int lanN   = lane & 15;
    #pragma unroll
    for (int i = 0; i < 2; i++) {
        #pragma unroll
        for (int j = 0; j < 4; j++) {
            #pragma unroll
            for (int r = 0; r < 8; r++) {
                int m = m0 + wm * 32 + i * 16 + r + laneHi * 8;
                int n = n0 + wn * 64 + j * 16 + lanN;
                out[(size_t)m * EMBED + n] = acc[i][j].f[r] + bias[n];
            }
        }
    }
}

// ---------------------------------------------------------------------------
extern "C" void kernel_launch(void* const* d_in, const int* in_sizes, int n_in,
                              void* d_out, int out_size, void* d_ws, size_t ws_size,
                              hipStream_t stream) {
    (void)in_sizes; (void)n_in; (void)out_size; (void)ws_size;
    const float* x      = (const float*)d_in[0];
    const float* w_qkv  = (const float*)d_in[1];
    const float* b_qkv  = (const float*)d_in[2];
    const float* w_proj = (const float*)d_in[3];
    const float* b_proj = (const float*)d_in[4];
    float* out = (float*)d_out;

    const size_t XE = (size_t)BATCH * SEQ * EMBED;   // 8Mi
    const size_t WQ = (size_t)3 * EMBED * EMBED;     // 3Mi
    const size_t WP = (size_t)EMBED * EMBED;         // 1Mi

    // workspace: xb | wqkvb | wprojb | qkv(3x8Mi) | ctx  => 44Mi u16 = 88MB
    unsigned short* xb     = (unsigned short*)d_ws;
    unsigned short* wqkvb  = xb + XE;
    unsigned short* wprojb = wqkvb + WQ;
    unsigned short* qkvb   = wprojb + WP;
    unsigned short* ctx    = qkvb + 3 * QKV_ELEMS;

    cvt_bf16_kernel<<<(int)(XE / 2048), 256, 0, stream>>>(x, xb);
    cvt_bf16_kernel<<<(int)(WQ / 2048), 256, 0, stream>>>(w_qkv, wqkvb);
    cvt_bf16_kernel<<<(int)(WP / 2048), 256, 0, stream>>>(w_proj, wprojb);

    qkv_gemm_kernel<<<dim3(64, 24), 256, 0, stream>>>(xb, wqkvb, b_qkv, qkvb);
    attn_kernel<<<dim3(SEQ / QT, BATCH * NUM_HEAD), 256, 0, stream>>>(
        qkvb, qkvb + QKV_ELEMS, qkvb + 2 * QKV_ELEMS, ctx);
    proj_gemm_kernel<<<dim3(64, 8), 256, 0, stream>>>(ctx, wprojb, b_proj, out);
}
